// DecoderAdap_AttModule_7636451852734
// MI455X (gfx1250) — matrix operand order repaired and checked by reading, not compiled, not run on hardware
//
#include <hip/hip_runtime.h>
#include <math.h>

// ---------------------------------------------------------------------------
// Types for CDNA5 WMMA (wave32)
// ---------------------------------------------------------------------------
typedef __bf16 bf16;
typedef __bf16 v16bf __attribute__((ext_vector_type(16)));
typedef float  v8f   __attribute__((ext_vector_type(8)));
typedef float  f32x4 __attribute__((ext_vector_type(4)));

#define BB   64      // batch
#define NREG 36
#define FDIM 2048
#define DD   1024
#define AA   512
#define EE   1024
#define VV   10000
#define VP   10048   // VV padded to multiple of 64
#define LL   20
#define TT   19      // L-1 decode steps

__device__ __forceinline__ float sigm(float x) { return 1.0f / (1.0f + expf(-x)); }

// ---------------------------------------------------------------------------
// bf16 WMMA GEMM:  C[M,Nout] = A[M,K] @ W[K,N] (+bias) (+=C) (row mask)
//   M multiple of 64 (grid.x), K multiple of 32, N (padded) multiple of 64,
//   Nout <= N logical width for bias/store.
//   8 waves/block, 64x64 tile, double-buffered LDS, 2x wmma/wave/k-step.
// ---------------------------------------------------------------------------
__global__ __launch_bounds__(256)
void gemm_bf16_k(const bf16* __restrict__ A, const bf16* __restrict__ W,
                 const float* __restrict__ bias, float* __restrict__ C,
                 int N, int Nout, int K, long ldc,
                 int accumulate, const int* __restrict__ rowmask, int tstep)
{
    __shared__ __align__(16) bf16 As[2][64 * 32];   // [m][k]
    __shared__ __align__(16) bf16 Bs[2][64 * 32];   // [n][k] (transposed)

    const int tid  = threadIdx.x;
    const int wv   = tid >> 5;          // 0..7
    const int lane = tid & 31;
    const int hv   = lane >> 4;         // 0/1 (lane half)
    const int l16  = lane & 15;
    const int bm0  = blockIdx.x * 64;
    const int bn0  = blockIdx.y * 64;
    const int mrow = (wv & 3) * 16;     // wave's M sub-tile
    const int ncol = (wv >> 2) * 32;    // wave's N sub-tile base

    const int ar = tid >> 2, ac = (tid & 3) * 8;   // A staging: row, col8
    const int bk = tid >> 3, bn8 = (tid & 7) * 8;  // B staging: k, n8

    const bf16* aPtr = A + (size_t)(bm0 + ar) * K + ac;
    const bf16* wPtr = W + (size_t)bk * N + bn0 + bn8;

    // prologue: stage k-tile 0 into buffer 0
    {
        f32x4 aR = *(const f32x4*)aPtr;
        f32x4 bR = *(const f32x4*)wPtr;
        *(f32x4*)&As[0][ar * 32 + ac] = aR;
        const bf16* tb = (const bf16*)&bR;
        #pragma unroll
        for (int i = 0; i < 8; ++i) Bs[0][(bn8 + i) * 32 + bk] = tb[i];
    }

    v8f acc0 = {}; v8f acc1 = {};
    const int nT = K >> 5;

    for (int it = 0; it < nT; ++it) {
        const int cur = it & 1;
        __syncthreads();                      // buffer[cur] ready

        // issue next tile's global loads early (overlap with WMMA below)
        f32x4 aR, bR;
        const bool more = (it + 1) < nT;
        if (more) {
            aR = *(const f32x4*)(aPtr + (it + 1) * 32);
            bR = *(const f32x4*)(wPtr + (size_t)(it + 1) * 32 * N);
            if (it + 2 < nT) {
                __builtin_prefetch(aPtr + (it + 2) * 32, 0, 1);
                __builtin_prefetch(wPtr + (size_t)(it + 2) * 32 * N, 0, 1);
            }
        }

        // Fragments per documented CDNA5 layouts.
        // A 16x32 bf16: lanes0-15 K{0..7,16..23}, lanes16-31 K{8..15,24..31}
        union { v16bf v; f32x4 q[2]; } fa, fb0, fb1;
        const bf16* ap = &As[cur][(mrow + l16) * 32 + hv * 8];
        fa.q[0] = *(const f32x4*)ap;
        fa.q[1] = *(const f32x4*)(ap + 16);
        // B 32x16 bf16: lanes0-15 N=l16,K{0..7,16..23} ; lanes16-31 N=l16,K{8..15,24..31}
        const bf16* bp0 = &Bs[cur][(ncol + l16) * 32 + hv * 8];
        fb0.q[0] = *(const f32x4*)bp0;
        fb0.q[1] = *(const f32x4*)(bp0 + 16);
        const bf16* bp1 = &Bs[cur][(ncol + 16 + l16) * 32 + hv * 8];
        fb1.q[0] = *(const f32x4*)bp1;
        fb1.q[1] = *(const f32x4*)(bp1 + 16);

        acc0 = __builtin_amdgcn_wmma_f32_16x16x32_bf16(
            false, fa.v, false, fb0.v, (short)0, acc0, false, false);
        acc1 = __builtin_amdgcn_wmma_f32_16x16x32_bf16(
            false, fa.v, false, fb1.v, (short)0, acc1, false, false);

        // stage next tile into the other buffer
        if (more) {
            *(f32x4*)&As[cur ^ 1][ar * 32 + ac] = aR;
            const bf16* tb = (const bf16*)&bR;
            #pragma unroll
            for (int i = 0; i < 8; ++i) Bs[cur ^ 1][(bn8 + i) * 32 + bk] = tb[i];
        }
    }

    // C/D layout: lane l16 = N; VGPR r -> M = r + 8*hv
    const int n0 = bn0 + ncol + l16;
    const int n1 = n0 + 16;
    const float b0 = (bias && n0 < Nout) ? bias[n0] : 0.0f;
    const float b1 = (bias && n1 < Nout) ? bias[n1] : 0.0f;
    #pragma unroll
    for (int r = 0; r < 8; ++r) {
        const int m = bm0 + mrow + hv * 8 + r;
        float* crow = C + (size_t)m * ldc;
        float v0 = acc0[r] + b0;
        float v1 = acc1[r] + b1;
        if (accumulate) {
            if (n0 < Nout) v0 += crow[n0];
            if (n1 < Nout) v1 += crow[n1];
        }
        if (rowmask && tstep >= rowmask[m]) { v0 = 0.0f; v1 = 0.0f; }
        if (n0 < Nout) crow[n0] = v0;
        if (n1 < Nout) crow[n1] = v1;
    }
}

// ---------------------------------------------------------------------------
// One-time prep kernels
// ---------------------------------------------------------------------------
__global__ void conv_bf16_k(const float* __restrict__ s, bf16* __restrict__ d, long n)
{
    long i = (long)blockIdx.x * 4096 + threadIdx.x;
    #pragma unroll
    for (int it = 0; it < 16; ++it) {
        long j = i + (long)it * 256;
        if (j < n) d[j] = (bf16)s[j];
    }
}

// copy rows x cols fp32 -> bf16 with padded row stride (zeros in pad)
__global__ void conv_pad_k(const float* __restrict__ s, bf16* __restrict__ d,
                           long rows, int cols, int stride)
{
    const long total = rows * (long)stride;
    long i = (long)blockIdx.x * 4096 + threadIdx.x;
    #pragma unroll
    for (int it = 0; it < 16; ++it) {
        long j = i + (long)it * 256;
        if (j < total) {
            long r = j / stride;
            int  c = (int)(j - r * stride);
            d[j] = (c < cols) ? (bf16)s[r * (long)cols + c] : (bf16)0.0f;
        }
    }
}

__global__ void vec_add_k(const float* a, const float* b, float* o, int n)
{
    int i = blockIdx.x * 256 + threadIdx.x;
    if (i < n) o[i] = a[i] + b[i];
}

// stable argsort of -sizes, gather seqs; also write int outputs
__global__ void sort_k(const int* __restrict__ sizes, const int* __restrict__ seqs,
                       int* __restrict__ ord, int* __restrict__ declen,
                       int* __restrict__ seqs_s, int* __restrict__ out_i)
{
    __shared__ int sz[BB];
    __shared__ int os[BB];
    const int i = threadIdx.x;
    sz[i] = sizes[i];
    __syncthreads();
    const int si = sz[i];
    int rank = 0;
    for (int j = 0; j < BB; ++j) {
        int sj = sz[j];
        if (sj > si || (sj == si && j < i)) rank++;
    }
    os[rank] = i;
    __syncthreads();
    const int src = os[i];
    ord[i] = src;
    out_i[BB * LL + BB + i] = src;            // order
    const int dl = sz[src] - 1;
    declen[i] = dl;
    out_i[BB * LL + i] = dl;                  // dec_len
    for (int l = 0; l < LL; ++l) {
        int v = seqs[src * LL + l];
        seqs_s[i * LL + l] = v;
        out_i[i * LL + l] = v;                // seqs (sorted)
    }
}

// feats average + reordered bf16 copy of feats (A matrix for img_att GEMM)
__global__ void prep_feats_k(const float* __restrict__ feats, const int* __restrict__ ord,
                             bf16* __restrict__ favg, bf16* __restrict__ Aimg)
{
    const int b = blockIdx.x;
    const float* fb = feats + (size_t)ord[b] * NREG * FDIM;
    for (int d = threadIdx.x; d < FDIM; d += 256) {
        float s = 0.0f;
        for (int r = 0; r < NREG; ++r) {
            float v = fb[r * FDIM + d];
            s += v;
            Aimg[((size_t)b * NREG + r) * FDIM + d] = (bf16)v;
        }
        favg[b * FDIM + d] = (bf16)(s * (1.0f / (float)NREG));
    }
}

// ---------------------------------------------------------------------------
// Per-step concat builders (bf16 activations)
// ---------------------------------------------------------------------------
__global__ void build_x1_k(bf16* __restrict__ X1, const bf16* __restrict__ h2b,
                           const bf16* __restrict__ favg, const float* __restrict__ emb,
                           const int* __restrict__ seqs_s, const bf16* __restrict__ h1b, int t)
{
    const int b = blockIdx.x;
    const int tok = seqs_s[b * LL + t];
    const float* e = emb + (size_t)tok * EE;
    bf16* x = X1 + (size_t)b * (DD + FDIM + EE + DD);
    for (int j = threadIdx.x; j < DD + FDIM + EE + DD; j += 256) {
        bf16 v;
        if (j < DD)                  v = h2b[b * DD + j];
        else if (j < DD + FDIM)      v = favg[b * FDIM + (j - DD)];
        else if (j < DD + FDIM + EE) v = (bf16)e[j - DD - FDIM];
        else                         v = h1b[b * DD + (j - DD - FDIM - EE)];
        x[j] = v;
    }
}

__global__ void build_xg_k(bf16* __restrict__ Xg, const float* __restrict__ emb,
                           const int* __restrict__ seqs_s, const bf16* __restrict__ h1b, int t)
{
    const int b = blockIdx.x;
    const int tok = seqs_s[b * LL + t];
    const float* e = emb + (size_t)tok * EE;
    bf16* x = Xg + (size_t)b * (EE + DD);
    for (int j = threadIdx.x; j < EE + DD; j += 256)
        x[j] = (j < EE) ? (bf16)e[j] : h1b[b * DD + (j - EE)];
}

__global__ void build_x2_k(bf16* __restrict__ X2, const float* __restrict__ AW,
                           const bf16* __restrict__ h1b, const bf16* __restrict__ h2b)
{
    const int b = blockIdx.x;
    bf16* x = X2 + (size_t)b * (FDIM + DD + DD);
    for (int j = threadIdx.x; j < FDIM + DD + DD; j += 256) {
        bf16 v;
        if (j < FDIM)           v = (bf16)AW[b * FDIM + j];
        else if (j < FDIM + DD) v = h1b[b * DD + (j - FDIM)];
        else                    v = h2b[b * DD + (j - FDIM - DD)];
        x[j] = v;
    }
}

__global__ void build_xp_k(bf16* __restrict__ Xp, const bf16* __restrict__ h2b,
                           const float* __restrict__ AW)
{
    const int b = blockIdx.x;
    bf16* x = Xp + (size_t)b * (DD + FDIM);
    for (int j = threadIdx.x; j < DD + FDIM; j += 256)
        x[j] = (j < DD) ? h2b[b * DD + j] : (bf16)AW[b * FDIM + (j - DD)];
}

// ---------------------------------------------------------------------------
// Pointwise kernels
// ---------------------------------------------------------------------------
__global__ void lstm_pw_k(const float* __restrict__ G, float* __restrict__ c,
                          float* __restrict__ h, bf16* __restrict__ hb)
{
    const int idx = blockIdx.x * 256 + threadIdx.x;   // 64*1024 total
    const int b = idx >> 10, j = idx & 1023;
    const float* g = G + (size_t)b * 4 * DD;
    const float i_ = sigm(g[j]);
    const float f_ = sigm(g[DD + j]);
    const float gg = tanhf(g[2 * DD + j]);
    const float o_ = sigm(g[3 * DD + j]);
    const float cn = f_ * c[idx] + i_ * gg;
    const float hn = o_ * tanhf(cn);
    c[idx] = cn;
    h[idx] = hn;
    hb[idx] = (bf16)hn;
}

__global__ void gate_pw_k(const float* __restrict__ Gg, const float* __restrict__ c1,
                          float* __restrict__ sent, bf16* __restrict__ sentb)
{
    const int idx = blockIdx.x * 256 + threadIdx.x;
    const float s = sigm(Gg[idx]) * tanhf(c1[idx]);
    sent[idx] = s;
    sentb[idx] = (bf16)s;
}

__global__ void sentbeta_pw_k(const float* __restrict__ sent, const float* __restrict__ beta,
                              bf16* __restrict__ sentbb)
{
    const int idx = blockIdx.x * 256 + threadIdx.x;
    sentbb[idx] = (bf16)(sent[idx] * beta[idx >> 10]);
}

// ---------------------------------------------------------------------------
// Adaptive attention: scores, softmax(37), weighted image sum (+bws), beta
// ---------------------------------------------------------------------------
__global__ __launch_bounds__(256)
void attention_k(const float* __restrict__ h1att, const float* __restrict__ sentatt,
                 const float* __restrict__ imgatt, const float* __restrict__ Wa,
                 const float* __restrict__ ba, const float* __restrict__ bws,
                 const float* __restrict__ feats, const int* __restrict__ ord,
                 float* __restrict__ AW, float* __restrict__ beta)
{
    __shared__ float sc[NREG + 1];
    const int b = blockIdx.x;
    const int wv = threadIdx.x >> 5;
    const int lane = threadIdx.x & 31;
    const float* hatt = h1att + (size_t)b * AA;

    for (int r = wv; r < NREG + 1; r += 8) {
        const float* base = (r < NREG) ? (imgatt + ((size_t)b * NREG + r) * AA)
                                       : (sentatt + (size_t)b * AA);
        float s = 0.0f;
        for (int j = lane; j < AA; j += 32) {
            float v = hatt[j] + base[j];
            v = v > 0.0f ? v : 0.0f;
            s += v * Wa[j];
        }
        #pragma unroll
        for (int o = 16; o > 0; o >>= 1) s += __shfl_down(s, o, 32);
        if (lane == 0) sc[r] = s + ba[0];
    }
    __syncthreads();
    if (threadIdx.x == 0) {
        float mx = -3.4e38f;
        for (int r = 0; r < NREG + 1; ++r) mx = fmaxf(mx, sc[r]);
        float sum = 0.0f;
        for (int r = 0; r < NREG + 1; ++r) { float e = expf(sc[r] - mx); sc[r] = e; sum += e; }
        float inv = 1.0f / sum;
        for (int r = 0; r < NREG + 1; ++r) sc[r] *= inv;
        beta[b] = sc[NREG];
    }
    __syncthreads();
    const float* fb = feats + (size_t)ord[b] * NREG * FDIM;
    for (int d = threadIdx.x; d < FDIM; d += 256) {
        float a = bws[d];
        for (int r = 0; r < NREG; ++r) a += sc[r] * fb[r * FDIM + d];
        AW[b * FDIM + d] = a;
    }
}

// ---------------------------------------------------------------------------
// Host side
// ---------------------------------------------------------------------------
extern "C" void kernel_launch(void* const* d_in, const int* in_sizes, int n_in,
                              void* d_out, int out_size, void* d_ws, size_t ws_size,
                              hipStream_t stream)
{
    (void)in_sizes; (void)n_in; (void)out_size; (void)ws_size;

    const float* feats   = (const float*)d_in[0];
    const float* emb     = (const float*)d_in[1];
    const float* td_Wih  = (const float*)d_in[2];
    const float* td_Whh  = (const float*)d_in[3];
    const float* td_bih  = (const float*)d_in[4];
    const float* td_bhh  = (const float*)d_in[5];
    const float* lg_Wih  = (const float*)d_in[6];
    const float* lg_Whh  = (const float*)d_in[7];
    const float* lg_bih  = (const float*)d_in[8];
    const float* lg_bhh  = (const float*)d_in[9];
    const float* Wge     = (const float*)d_in[10];
    const float* bge     = (const float*)d_in[11];
    const float* Wgh     = (const float*)d_in[12];
    const float* bgh     = (const float*)d_in[13];
    const float* Wf      = (const float*)d_in[14];
    const float* bf_     = (const float*)d_in[15];
    const float* Wd      = (const float*)d_in[16];
    const float* bd      = (const float*)d_in[17];
    const float* Ws      = (const float*)d_in[18];
    const float* bs      = (const float*)d_in[19];
    const float* Wws     = (const float*)d_in[20];
    const float* bws     = (const float*)d_in[21];
    const float* Wa      = (const float*)d_in[22];
    const float* ba      = (const float*)d_in[23];
    const float* Wh      = (const float*)d_in[24];
    const float* bh      = (const float*)d_in[25];
    const float* Wi      = (const float*)d_in[26];
    const float* bi      = (const float*)d_in[27];
    const int* sequences = (const int*)d_in[28];
    const int* sizes     = (const int*)d_in[29];

    // --- workspace carving ---
    char* wsb = (char*)d_ws;
    size_t off = 0;
    auto carve = [&](size_t bytes) -> void* {
        off = (off + 255) & ~(size_t)255;
        void* p = wsb + off;
        off += bytes;
        return p;
    };
    int*  ord     = (int*)carve(BB * 4);
    int*  declen  = (int*)carve(BB * 4);
    int*  seqs_s  = (int*)carve(BB * LL * 4);
    bf16* favg    = (bf16*)carve((size_t)BB * FDIM * 2);
    bf16* Aimg    = (bf16*)carve((size_t)BB * NREG * FDIM * 2);
    float* imgatt = (float*)carve((size_t)BB * NREG * AA * 4);
    bf16* Wtd     = (bf16*)carve((size_t)5120 * 4096 * 2);   // [td_Wih ; td_Whh]
    bf16* Wlg     = (bf16*)carve((size_t)4096 * 4096 * 2);   // [lg_Wih ; lg_Whh]
    bf16* Wg      = (bf16*)carve((size_t)2048 * 1024 * 2);   // [Wge ; Wgh]
    bf16* Wdb     = (bf16*)carve((size_t)1024 * 512 * 2);
    bf16* Wsb     = (bf16*)carve((size_t)1024 * 512 * 2);
    bf16* Wwsb    = (bf16*)carve((size_t)1024 * 2048 * 2);
    bf16* Wpb     = (bf16*)carve((size_t)3072 * VP * 2);     // [Wh ; Wi], N padded
    bf16* Wfb     = (bf16*)carve((size_t)2048 * 512 * 2);
    float* btd    = (float*)carve(4096 * 4);
    float* blg    = (float*)carve(4096 * 4);
    float* bg     = (float*)carve(1024 * 4);
    float* bp     = (float*)carve(VV * 4);
    bf16* X1      = (bf16*)carve((size_t)BB * 5120 * 2);
    bf16* Xg      = (bf16*)carve((size_t)BB * 2048 * 2);
    bf16* X2      = (bf16*)carve((size_t)BB * 4096 * 2);
    bf16* Xp      = (bf16*)carve((size_t)BB * 3072 * 2);
    float* G1     = (float*)carve((size_t)BB * 4096 * 4);
    float* Gg     = (float*)carve((size_t)BB * 1024 * 4);
    float* G2     = (float*)carve((size_t)BB * 4096 * 4);
    float* h1     = (float*)carve((size_t)BB * DD * 4);
    float* c1     = (float*)carve((size_t)BB * DD * 4);
    float* h2     = (float*)carve((size_t)BB * DD * 4);
    float* c2     = (float*)carve((size_t)BB * DD * 4);
    bf16* h1b     = (bf16*)carve((size_t)BB * DD * 2);
    bf16* h2b     = (bf16*)carve((size_t)BB * DD * 2);
    float* sent   = (float*)carve((size_t)BB * DD * 4);
    bf16* sentb   = (bf16*)carve((size_t)BB * DD * 2);
    bf16* sentbb  = (bf16*)carve((size_t)BB * DD * 2);
    float* h1att  = (float*)carve((size_t)BB * AA * 4);
    float* sentatt= (float*)carve((size_t)BB * AA * 4);
    float* beta   = (float*)carve(BB * 4);
    float* AW     = (float*)carve((size_t)BB * FDIM * 4);

    float* preds = (float*)d_out;                                  // [B][T][V]
    int* out_i = (int*)((float*)d_out + (size_t)BB * TT * VV);     // seqs, dec_len, order

    auto conv = [&](const float* s, bf16* d, long n) {
        conv_bf16_k<<<(unsigned)((n + 4095) / 4096), 256, 0, stream>>>(s, d, n);
    };
    auto convp = [&](const float* s, bf16* d, long rows, int cols, int stride) {
        long n = rows * (long)stride;
        conv_pad_k<<<(unsigned)((n + 4095) / 4096), 256, 0, stream>>>(s, d, rows, cols, stride);
    };
    auto gemm = [&](const bf16* A, const bf16* W, const float* bias, float* C,
                    int M, int N, int Nout, int K, long ldc, int accum,
                    const int* rowmask, int t) {
        dim3 g((unsigned)(M / 64), (unsigned)(N / 64));
        gemm_bf16_k<<<g, 256, 0, stream>>>(A, W, bias, C, N, Nout, K, ldc,
                                           accum, rowmask, t);
    };

    // ---- one-time prep ----
    sort_k<<<1, BB, 0, stream>>>(sizes, sequences, ord, declen, seqs_s, out_i);

    conv(td_Wih, Wtd,                     (long)4096 * 4096);
    conv(td_Whh, Wtd + (size_t)4096*4096, (long)1024 * 4096);
    conv(lg_Wih, Wlg,                     (long)3072 * 4096);
    conv(lg_Whh, Wlg + (size_t)3072*4096, (long)1024 * 4096);
    conv(Wge,    Wg,                      (long)1024 * 1024);
    conv(Wgh,    Wg + (size_t)1024*1024,  (long)1024 * 1024);
    conv(Wd,     Wdb,                     (long)1024 * 512);
    conv(Ws,     Wsb,                     (long)1024 * 512);
    conv(Wws,    Wwsb,                    (long)1024 * 2048);
    convp(Wh,    Wpb,                     1024, VV, VP);
    convp(Wi,    Wpb + (size_t)1024*VP,   2048, VV, VP);
    conv(Wf,     Wfb,                     (long)2048 * 512);

    vec_add_k<<<16, 256, 0, stream>>>(td_bih, td_bhh, btd, 4096);
    vec_add_k<<<16, 256, 0, stream>>>(lg_bih, lg_bhh, blg, 4096);
    vec_add_k<<<4, 256, 0, stream>>>(bge, bgh, bg, 1024);
    vec_add_k<<<(VV + 255) / 256, 256, 0, stream>>>(bh, bi, bp, VV);

    prep_feats_k<<<BB, 256, 0, stream>>>(feats, ord, favg, Aimg);

    // loop-invariant: img_att = feats_s @ Wf + bf   (M=2304, N=512, K=2048)
    gemm(Aimg, Wfb, bf_, imgatt, BB * NREG, AA, AA, FDIM, AA, 0, nullptr, 0);

    // zero initial states
    hipMemsetAsync(h1, 0, (size_t)BB * DD * 4, stream);
    hipMemsetAsync(c1, 0, (size_t)BB * DD * 4, stream);
    hipMemsetAsync(h2, 0, (size_t)BB * DD * 4, stream);
    hipMemsetAsync(c2, 0, (size_t)BB * DD * 4, stream);
    hipMemsetAsync(h1b, 0, (size_t)BB * DD * 2, stream);
    hipMemsetAsync(h2b, 0, (size_t)BB * DD * 2, stream);

    // ---- decode loop ----
    for (int t = 0; t < TT; ++t) {
        // top-down LSTM: x1 = [h2, favg, e_t, h1],  W = [td_Wih ; td_Whh]
        build_x1_k<<<BB, 256, 0, stream>>>(X1, h2b, favg, emb, seqs_s, h1b, t);
        gemm(X1, Wtd, btd, G1, BB, 4096, 4096, 5120, 4096, 0, nullptr, 0);
        lstm_pw_k<<<BB * DD / 256, 256, 0, stream>>>(G1, c1, h1, h1b);

        // gate: [e_t, h1] @ [Wge ; Wgh]
        build_xg_k<<<BB, 256, 0, stream>>>(Xg, emb, seqs_s, h1b, t);
        gemm(Xg, Wg, bg, Gg, BB, 1024, 1024, 2048, 1024, 0, nullptr, 0);
        gate_pw_k<<<BB * DD / 256, 256, 0, stream>>>(Gg, c1, sent, sentb);

        // attention projections
        gemm(h1b,   Wdb, bd, h1att,   BB, AA, AA, DD, AA, 0, nullptr, 0);
        gemm(sentb, Wsb, bs, sentatt, BB, AA, AA, DD, AA, 0, nullptr, 0);

        // attention + weighted image sum (+bws), then AW += (sent*beta)@Wws
        attention_k<<<BB, 256, 0, stream>>>(h1att, sentatt, imgatt, Wa, ba, bws,
                                            feats, ord, AW, beta);
        sentbeta_pw_k<<<BB * DD / 256, 256, 0, stream>>>(sent, beta, sentbb);
        gemm(sentbb, Wwsb, nullptr, AW, BB, FDIM, FDIM, DD, FDIM, 1, nullptr, 0);

        // language LSTM: x2 = [aw, h1, h2],  W = [lg_Wih ; lg_Whh]
        build_x2_k<<<BB, 256, 0, stream>>>(X2, AW, h1b, h2b);
        gemm(X2, Wlg, blg, G2, BB, 4096, 4096, 4096, 4096, 0, nullptr, 0);
        lstm_pw_k<<<BB * DD / 256, 256, 0, stream>>>(G2, c2, h2, h2b);

        // prediction: [h2, aw] @ [Wh ; Wi] + (bh+bi), masked, into preds[:,t,:]
        build_xp_k<<<BB, 256, 0, stream>>>(Xp, h2b, AW);
        gemm(Xp, Wpb, bp, preds + (size_t)t * VV, BB, VP, VV, 3072,
             (long)TT * VV, 0, declen, t);
    }
}
